// CausalSelfAttention_44839458570342
// MI455X (gfx1250) — matrix operand order repaired and checked by reading, not compiled, not run on hardware
//
#include <hip/hip_runtime.h>
#include <hip/hip_bf16.h>
#include <math.h>

// ---------------------------------------------------------------- constants
constexpr int Bb = 4;
constexpr int Tt = 2048;
constexpr int Cc = 2048;
constexpr int Hh = 16;
constexpr int Dh = 128;
constexpr int Mrows = Bb * Tt;            // 8192
constexpr float RMS_EPS = 1.1920929e-7f;  // FLT_EPSILON, matches jnp.finfo(f32).eps
constexpr float LOG_ROPE_BASE = 9.210340371976184f; // ln(10000)

typedef __bf16 bf16;
typedef __attribute__((ext_vector_type(16))) __bf16 v16bf;
typedef __attribute__((ext_vector_type(8)))  __bf16 bf16x8;
typedef __attribute__((ext_vector_type(8)))  float  v8f;
typedef __attribute__((ext_vector_type(4)))  unsigned int v4u;
typedef __attribute__((ext_vector_type(4)))  int v4i_t;
typedef __attribute__((ext_vector_type(8)))  int v8i_t;

#if __has_builtin(__builtin_amdgcn_tensor_load_to_lds) && \
    __has_builtin(__builtin_amdgcn_s_wait_tensorcnt)
#define USE_TDM 1
#else
#define USE_TDM 0
#endif

// Build a 16-element bf16 A/B fragment from two 16-byte LDS/global chunks.
__device__ __forceinline__ v16bf make_frag(const bf16* lo, const bf16* hi) {
  bf16x8 a = *(const bf16x8*)lo;
  bf16x8 b = *(const bf16x8*)hi;
  v16bf r;
#pragma unroll
  for (int i = 0; i < 8; ++i) { r[i] = a[i]; r[i + 8] = b[i]; }
  return r;
}

__device__ __forceinline__ v8f wmma_bf16(v16bf a, v16bf b, v8f c) {
  // D = A(16x32 bf16) * B(32x16 bf16) + C(16x16 f32)
  return __builtin_amdgcn_wmma_f32_16x16x32_bf16(false, a, false, b, (short)0, c,
                                                 false, false);
}

// ---------------------------------------------------------------- kernel 1:
// row RMS-norm over C, fp32 -> bf16
__global__ __launch_bounds__(256) void rmsnorm_rows_kernel(
    const float* __restrict__ x, bf16* __restrict__ out) {
  __shared__ float red[8];
  const size_t row = blockIdx.x;
  const float* xr = x + row * (size_t)Cc;
  const int tid = threadIdx.x, wave = tid >> 5, lane = tid & 31;
  const int base = tid * 8;
  float4 v0 = *(const float4*)(xr + base);
  float4 v1 = *(const float4*)(xr + base + 4);
  float s = v0.x * v0.x + v0.y * v0.y + v0.z * v0.z + v0.w * v0.w +
            v1.x * v1.x + v1.y * v1.y + v1.z * v1.z + v1.w * v1.w;
#pragma unroll
  for (int off = 16; off >= 1; off >>= 1) s += __shfl_xor(s, off);
  if (lane == 0) red[wave] = s;
  __syncthreads();
  float tot = 0.f;
#pragma unroll
  for (int w = 0; w < 8; ++w) tot += red[w];
  const float r = rsqrtf(tot / (float)Cc + RMS_EPS);
  bf16* o = out + row * (size_t)Cc + base;
  o[0] = (bf16)(v0.x * r); o[1] = (bf16)(v0.y * r);
  o[2] = (bf16)(v0.z * r); o[3] = (bf16)(v0.w * r);
  o[4] = (bf16)(v1.x * r); o[5] = (bf16)(v1.y * r);
  o[6] = (bf16)(v1.z * r); o[7] = (bf16)(v1.w * r);
}

// ---------------------------------------------------------------- kernel 2:
// fp32 -> bf16 convert (weights)
__global__ __launch_bounds__(256) void cvt_kernel(const float* __restrict__ in,
                                                  bf16* __restrict__ out,
                                                  size_t n) {
  size_t i = ((size_t)blockIdx.x * 256 + threadIdx.x) * 4;
  if (i + 3 < n) {
    float4 v = *(const float4*)(in + i);
    out[i + 0] = (bf16)v.x; out[i + 1] = (bf16)v.y;
    out[i + 2] = (bf16)v.z; out[i + 3] = (bf16)v.w;
  }
}

// ---------------------------------------------------------------- kernel 3:
// GEMM: out[m,n] = sum_k A[m,k] * W[n,k]   (A: MxK bf16, W: NxK bf16)
// optional +residual, optional bf16 output. 128x128x64 tiles, 8 waves,
// each wave computes a 64x32 sub-tile as 4x2 WMMA 16x16 tiles.
#define GBM 128
#define GBN 128
#define GBK 64
#define GLD 72  // padded LDS row stride (elems): 144B, 16B aligned, conflict-free

__global__ __launch_bounds__(256) void gemm_bf16_kernel(
    const bf16* __restrict__ A, const bf16* __restrict__ W,
    void* __restrict__ out, const float* __restrict__ addend,
    int M, int N, int K, int out_bf16) {
  __shared__ bf16 As[GBM * GLD];
  __shared__ bf16 Bs[GBN * GLD];
  const int tid = threadIdx.x;
  const int wave = tid >> 5, lane = tid & 31;
  const int half = lane >> 4, l16 = lane & 15;
  const int waveM = wave >> 2, waveN = wave & 3;
  const int gm0 = blockIdx.y * GBM;
  const int gn0 = blockIdx.x * GBN;

  v8f acc[4][2];
#pragma unroll
  for (int mt = 0; mt < 4; ++mt)
#pragma unroll
    for (int nt = 0; nt < 2; ++nt) acc[mt][nt] = (v8f)(0.f);

  const int lrow = tid >> 1;          // 0..127
  const int lcol = (tid & 1) * 32;    // 0 or 32

  for (int k0 = 0; k0 < K; k0 += GBK) {
    const bf16* ga = A + (size_t)(gm0 + lrow) * K + k0 + lcol;
    const bf16* gb = W + (size_t)(gn0 + lrow) * K + k0 + lcol;
    __builtin_prefetch(ga + 2 * GBK, 0, 3);   // global_prefetch_b8
    __builtin_prefetch(gb + 2 * GBK, 0, 3);
#pragma unroll
    for (int j = 0; j < 4; ++j) {
      *(bf16x8*)&As[lrow * GLD + lcol + 8 * j] = *(const bf16x8*)(ga + 8 * j);
      *(bf16x8*)&Bs[lrow * GLD + lcol + 8 * j] = *(const bf16x8*)(gb + 8 * j);
    }
    __syncthreads();

#pragma unroll
    for (int kc = 0; kc < 2; ++kc) {
      v16bf bfrag[2];
#pragma unroll
      for (int nt = 0; nt < 2; ++nt) {
        const bf16* p =
            &Bs[(waveN * 32 + nt * 16 + l16) * GLD + kc * 32];
        bfrag[nt] = make_frag(p + 8 * half, p + 16 + 8 * half);
      }
#pragma unroll
      for (int mt = 0; mt < 4; ++mt) {
        const bf16* p = &As[(waveM * 64 + mt * 16 + l16) * GLD + kc * 32];
        v16bf af = make_frag(p + 8 * half, p + 16 + 8 * half);
#pragma unroll
        for (int nt = 0; nt < 2; ++nt)
          acc[mt][nt] = wmma_bf16(af, bfrag[nt], acc[mt][nt]);
      }
    }
    __syncthreads();
  }

#pragma unroll
  for (int mt = 0; mt < 4; ++mt)
#pragma unroll
    for (int nt = 0; nt < 2; ++nt)
#pragma unroll
      for (int i = 0; i < 8; ++i) {
        const int r = gm0 + waveM * 64 + mt * 16 + half * 8 + i;
        const int c = gn0 + waveN * 32 + nt * 16 + l16;
        float v = acc[mt][nt][i];
        if (addend) v += addend[(size_t)r * N + c];
        if (out_bf16) ((bf16*)out)[(size_t)r * N + c] = (bf16)v;
        else          ((float*)out)[(size_t)r * N + c] = v;
      }
}

// ---------------------------------------------------------------- kernel 4:
// per-head RMS-norm + RoPE + scale, fp32 -> bf16.  One block per (b,t) row;
// each wave handles 2 heads of 128 elems (4 per lane), shuffle reductions.
__global__ __launch_bounds__(256) void rmsrope_kernel(
    const float* __restrict__ in, bf16* __restrict__ out, float outScale) {
  const size_t row = blockIdx.x;          // b*T + t
  const int t = (int)(row % Tt);
  const int tid = threadIdx.x, wave = tid >> 5, lane = tid & 31;
#pragma unroll
  for (int hh = 0; hh < 2; ++hh) {
    const int h = wave * 2 + hh;
    const float* xr = in + row * (size_t)Cc + h * Dh + lane * 4;
    float4 xv = *(const float4*)xr;
    float ss = xv.x * xv.x + xv.y * xv.y + xv.z * xv.z + xv.w * xv.w;
#pragma unroll
    for (int off = 16; off >= 1; off >>= 1) ss += __shfl_xor(ss, off);
    const float r = rsqrtf(ss / (float)Dh + RMS_EPS);
    float xn[4] = {xv.x * r, xv.y * r, xv.z * r, xv.w * r};
    bf16* o = out + row * (size_t)Cc + h * Dh + lane * 4;
#pragma unroll
    for (int jj = 0; jj < 4; ++jj) {
      const int i = lane * 4 + jj;
      const float other = __shfl_xor(xn[jj], 16);  // pairs (i, i+64)
      const int j = i & 63;
      const float invf = __expf((float)j * (-LOG_ROPE_BASE / 64.f));
      float sn, cs;
      __sincosf((float)t * invf, &sn, &cs);
      const float y = (i < 64) ? (xn[jj] * cs + other * sn)
                               : (-other * sn + xn[jj] * cs);
      o[jj] = (bf16)(y * outScale);
    }
  }
}

// ---------------------------------------------------------------- kernel 5:
// flash attention (causal), bf16 WMMA, f32 online softmax.
// block = 256 thr = 8 waves; block owns 128 q rows (16/wave) of one (b,h);
// K tile (64 rows) staged via the Tensor Data Mover (TDM) with hardware LDS
// padding; V tile transposed manually at stage time.
#define AKLD 136  // K-tile row stride: 64 x 128 (+8 pad elems = 4 DW) -> 272B
#define AVLD 72   // V^T row stride:   128 x  64 (+8 pad) -> 144B rows
#define APLD 72   // P scratch stride:  16 x  64 (+8 pad)

__global__ __launch_bounds__(256) void attn_kernel(
    const bf16* __restrict__ Q, const bf16* __restrict__ Kb,
    const bf16* __restrict__ Vb, bf16* __restrict__ Y) {
  __shared__ bf16 Ks[64 * AKLD];
  __shared__ bf16 VTs[128 * AVLD];
  __shared__ bf16 Ps[8 * 16 * APLD];

  const int tid = threadIdx.x;
  const int wave = tid >> 5, lane = tid & 31;
  const int half = lane >> 4, l16 = lane & 15;
  const int bh = blockIdx.y, b = bh / Hh, h = bh % Hh;
  const int q0 = blockIdx.x * 128;
  const int qrow0 = q0 + wave * 16;

  // Q fragments (held in registers for the whole kernel): 4 K-chunks of 32
  v16bf qf[4];
  {
    const size_t qbase = ((size_t)(b * Tt + qrow0 + l16) * Hh + h) * Dh;
#pragma unroll
    for (int kc = 0; kc < 4; ++kc) {
      const bf16* p = Q + qbase + kc * 32;
      qf[kc] = make_frag(p + 8 * half, p + 16 + 8 * half);
    }
  }

  v8f O[8];
#pragma unroll
  for (int dt = 0; dt < 8; ++dt) O[dt] = (v8f)(0.f);
  float m_i[8], l_i[8];
#pragma unroll
  for (int i = 0; i < 8; ++i) { m_i[i] = -1e30f; l_i[i] = 0.f; }

  const int kend = min(Tt, q0 + 128);
  for (int kk0 = 0; kk0 < kend; kk0 += 64) {
    // ---- stage K tile into LDS
#if USE_TDM
    if (wave == 0) {
      // TDM descriptor (ISA 8.3/8.4): 2D tile 64 rows x 128 bf16, row stride
      // C elems; HW pads LDS dest every 64 DW (256B row) with 4 DW -> AKLD.
      const unsigned long long gaddr = (unsigned long long)(uintptr_t)(
          Kb + ((size_t)(b * Tt + kk0) * Hh + h) * Dh);
      const unsigned int ldsb = (unsigned int)(uintptr_t)(&Ks[0]);
      v4u g0;
      g0[0] = 1u;                                   // count=1, user mode
      g0[1] = ldsb;                                 // lds_addr
      g0[2] = (unsigned int)gaddr;                  // global_addr[31:0]
      g0[3] = ((unsigned int)(gaddr >> 32) & 0x01FFFFFFu) | (2u << 30); // type=2
      v8i_t g1;
      g1[0] = (int)((1u << 16)        // data_size = 1 -> 2 bytes
                    | (1u << 20)      // pad_enable
                    | (5u << 22)      // pad_interval: 64 DWORDs
                    | (3u << 25));    // pad_amount:   4 DWORDs
      g1[1] = (int)(128u << 16);      // tensor_dim0 = 128 (low16 in [63:48])
      g1[2] = (int)(64u << 16);       // tensor_dim0 hi=0 | tensor_dim1 = 64
      g1[3] = (int)(128u << 16);      // tensor_dim1 hi=0 | tile_dim0 = 128
      g1[4] = 64;                     // tile_dim1 = 64, tile_dim2 = 0
      g1[5] = Cc;                     // tensor_dim0_stride = 2048 elems
      g1[6] = 0;
      g1[7] = 0;
      v4i_t gz = {0, 0, 0, 0};
#if defined(__clang_major__) && __clang_major__ >= 23
      v8i_t gz8 = {0, 0, 0, 0, 0, 0, 0, 0};
      __builtin_amdgcn_tensor_load_to_lds(g0, g1, gz, gz, gz8, 0);
#else
      __builtin_amdgcn_tensor_load_to_lds(g0, g1, gz, gz, 0);
#endif
      __builtin_amdgcn_s_wait_tensorcnt(0);
    }
#endif
    // ---- stage V tile transposed into LDS (and K manually if no TDM)
    {
      const int r = tid >> 2;              // 0..63 key row in tile
      const int c0 = (tid & 3) * 32;       // 0/32/64/96 d-offset
      const size_t gb = ((size_t)(b * Tt + kk0 + r) * Hh + h) * Dh + c0;
#if !USE_TDM
#pragma unroll
      for (int j = 0; j < 4; ++j)
        *(bf16x8*)&Ks[r * AKLD + c0 + 8 * j] = *(const bf16x8*)(Kb + gb + 8 * j);
#endif
      bf16 vtmp[32];
#pragma unroll
      for (int j = 0; j < 4; ++j)
        *(bf16x8*)&vtmp[8 * j] = *(const bf16x8*)(Vb + gb + 8 * j);
#pragma unroll
      for (int e = 0; e < 32; ++e) VTs[(c0 + e) * AVLD + r] = vtmp[e];
    }
    __syncthreads();

    // ---- S = Q * K^T  (scale already folded into Q)
    v8f S[4];
#pragma unroll
    for (int nt = 0; nt < 4; ++nt) {
      S[nt] = (v8f)(0.f);
#pragma unroll
      for (int kc = 0; kc < 4; ++kc) {
        const bf16* p = &Ks[(nt * 16 + l16) * AKLD + kc * 32];
        S[nt] = wmma_bf16(qf[kc], make_frag(p + 8 * half, p + 16 + 8 * half), S[nt]);
      }
    }

    // ---- causal mask + online softmax (rows live in 16-lane halves)
    float mnew[8];
#pragma unroll
    for (int i = 0; i < 8; ++i) {
      const int qg = qrow0 + half * 8 + i;
      float mx = m_i[i];
#pragma unroll
      for (int nt = 0; nt < 4; ++nt) {
        const int kg = kk0 + nt * 16 + l16;
        float s = (kg <= qg) ? S[nt][i] : -1e30f;
        S[nt][i] = s;
        mx = fmaxf(mx, s);
      }
#pragma unroll
      for (int off = 8; off >= 1; off >>= 1) mx = fmaxf(mx, __shfl_xor(mx, off));
      mnew[i] = mx;
    }
#pragma unroll
    for (int i = 0; i < 8; ++i) {
      const float alpha = __expf(m_i[i] - mnew[i]);
      m_i[i] = mnew[i];
      float s = 0.f;
#pragma unroll
      for (int nt = 0; nt < 4; ++nt) {
        const float p = __expf(S[nt][i] - mnew[i]);
        S[nt][i] = p;
        s += p;
      }
#pragma unroll
      for (int off = 8; off >= 1; off >>= 1) s += __shfl_xor(s, off);
      l_i[i] = l_i[i] * alpha + s;
#pragma unroll
      for (int dt = 0; dt < 8; ++dt) O[dt][i] *= alpha;
    }

    // ---- P (C-layout f32) -> bf16 via wave-private LDS (relayout to A)
    bf16* pw = &Ps[wave * 16 * APLD];
#pragma unroll
    for (int nt = 0; nt < 4; ++nt)
#pragma unroll
      for (int i = 0; i < 8; ++i)
        pw[(half * 8 + i) * APLD + nt * 16 + l16] = (bf16)S[nt][i];
    // same-wave LDS ops are in-order: no barrier needed before re-reading pw

    // ---- O += P * V   (V^T rows give contiguous B fragments)
#pragma unroll
    for (int c = 0; c < 2; ++c) {
      const bf16* pp = &pw[l16 * APLD + c * 32];
      const v16bf pf = make_frag(pp + 8 * half, pp + 16 + 8 * half);
#pragma unroll
      for (int dt = 0; dt < 8; ++dt) {
        const bf16* vp = &VTs[(dt * 16 + l16) * AVLD + c * 32];
        O[dt] = wmma_bf16(pf, make_frag(vp + 8 * half, vp + 16 + 8 * half), O[dt]);
      }
    }
    __syncthreads();  // protect Ks/VTs before next stage
  }

  // ---- finalize: O /= l, write bf16 y in (B,T,H,DH) layout
#pragma unroll
  for (int i = 0; i < 8; ++i) {
    const float inv = 1.f / l_i[i];
    const int qg = qrow0 + half * 8 + i;
#pragma unroll
    for (int dt = 0; dt < 8; ++dt)
      Y[((size_t)(b * Tt + qg) * Hh + h) * Dh + dt * 16 + l16] =
          (bf16)(O[dt][i] * inv);
  }
}

// ---------------------------------------------------------------- launch
static inline size_t alignup(size_t x) { return (x + 255) & ~(size_t)255; }

extern "C" void kernel_launch(void* const* d_in, const int* in_sizes, int n_in,
                              void* d_out, int out_size, void* d_ws,
                              size_t ws_size, hipStream_t stream) {
  const float* residual = (const float*)d_in[0];
  const float* wq = (const float*)d_in[1];
  const float* wk = (const float*)d_in[2];
  const float* wv = (const float*)d_in[3];
  const float* wp = (const float*)d_in[4];

  // workspace carve-up
  char* w = (char*)d_ws;
  const size_t nBT_C = (size_t)Mrows * Cc;
  const size_t nC_C = (size_t)Cc * Cc;
  bf16* xb = (bf16*)w;  w += alignup(nBT_C * 2);
  bf16* wqb = (bf16*)w; w += alignup(nC_C * 2);
  bf16* wkb = (bf16*)w; w += alignup(nC_C * 2);
  bf16* wvb = (bf16*)w; w += alignup(nC_C * 2);
  bf16* wpb = (bf16*)w; w += alignup(nC_C * 2);
  float* qf = (float*)w; w += alignup(nBT_C * 4);
  float* kf = (float*)w; w += alignup(nBT_C * 4);
  bf16* qb = (bf16*)w;  w += alignup(nBT_C * 2);
  bf16* kb = (bf16*)w;  w += alignup(nBT_C * 2);
  bf16* vb = (bf16*)w;  w += alignup(nBT_C * 2);
  bf16* yb = (bf16*)w;  w += alignup(nBT_C * 2);

  // 1) RMS-norm rows -> bf16 activations
  rmsnorm_rows_kernel<<<Mrows, 256, 0, stream>>>(residual, xb);

  // 2) weights fp32 -> bf16 (L2-resident afterwards: 192MB L2 >> 32MB)
  const int cvtGrid = (int)(nC_C / (256 * 4));
  cvt_kernel<<<cvtGrid, 256, 0, stream>>>(wq, wqb, nC_C);
  cvt_kernel<<<cvtGrid, 256, 0, stream>>>(wk, wkb, nC_C);
  cvt_kernel<<<cvtGrid, 256, 0, stream>>>(wv, wvb, nC_C);
  cvt_kernel<<<cvtGrid, 256, 0, stream>>>(wp, wpb, nC_C);

  // 3) QKV projections (WMMA bf16). v goes straight to bf16; q,k keep f32
  //    for the per-head RMS-norm.
  dim3 ggrid(Cc / GBN, Mrows / GBM);  // (16, 64)
  gemm_bf16_kernel<<<ggrid, 256, 0, stream>>>(xb, wqb, qf, nullptr,
                                              Mrows, Cc, Cc, 0);
  gemm_bf16_kernel<<<ggrid, 256, 0, stream>>>(xb, wkb, kf, nullptr,
                                              Mrows, Cc, Cc, 0);
  gemm_bf16_kernel<<<ggrid, 256, 0, stream>>>(xb, wvb, vb, nullptr,
                                              Mrows, Cc, Cc, 1);

  // 4) per-head RMS-norm + RoPE; fold softmax scale 1/sqrt(DH) into q
  const float qscale = 1.f / sqrtf((float)Dh);
  rmsrope_kernel<<<Mrows, 256, 0, stream>>>(qf, qb, qscale);
  rmsrope_kernel<<<Mrows, 256, 0, stream>>>(kf, kb, 1.0f);

  // 5) causal flash attention (TDM-staged K tiles)
  dim3 agrid(Tt / 128, Bb * Hh);  // (16, 64)
  attn_kernel<<<agrid, 256, 0, stream>>>(qb, kb, vb, yb);

  // 6) output projection + residual add -> fp32 d_out
  gemm_bf16_kernel<<<ggrid, 256, 0, stream>>>(yb, wpb, d_out, residual,
                                              Mrows, Cc, Cc, 0);
}
